// KnowledgeEmbedding_33071248179455
// MI455X (gfx1250) — compile-verified, matrix operand order checked by reading, not run on hardware
//
#include <hip/hip_runtime.h>

typedef __bf16 bf16_t;
typedef bf16_t v16bf __attribute__((ext_vector_type(16)));
typedef float  v8f   __attribute__((ext_vector_type(8)));

#define BATCH   16384
#define EMBED   128
#define NUM_NEG 100
#define NTILES  7              // ceil(100/16) N-tiles of negatives
#define KCHUNKS 4              // 128 / 32 (K per bf16 WMMA)
#define WAVES_PER_BLOCK 8
#define TILES   (BATCH / 16)   // 1024 M-tiles
#define BLOCKS  (TILES / WAVES_PER_BLOCK)  // 128

__device__ __forceinline__ float softplusf(float x) {
    // log(1 + exp(x)) = max(x,0) + log1p(exp(-|x|))  (numerically stable)
    return fmaxf(x, 0.0f) + log1pf(__expf(-fabsf(x)));
}

__global__ __launch_bounds__(256)
void ke_main_kernel(const float* __restrict__ head_table,
                    const float* __restrict__ tail_table,
                    const float* __restrict__ relation_vec,
                    const float* __restrict__ bias_table,
                    const int*   __restrict__ head_idxs,
                    const int*   __restrict__ tail_idxs,
                    const int*   __restrict__ neg_idxs,
                    float*       __restrict__ block_partials)
{
    // B operand slots: [ntile][kchunk][lane][16 bf16], operand-ready layout.
    __shared__ __align__(32) bf16_t sB[NTILES * KCHUNKS * 512];
    __shared__ float sWave[WAVES_PER_BLOCK];

    const int tid  = threadIdx.x;
    const int lane = tid & 31;
    const int wave = tid >> 5;
    const int m16  = lane & 15;   // row/col index within 16-tile
    const int hi16 = lane >> 4;   // which K half-group this lane holds

    // ---------------- Stage shared negatives into LDS (bf16, B layout) -----
    // B[k][n] = tail_table[neg_idxs[n]][k]; lane l holds col n=l%16,
    // K offsets: {32c + 8*(l/16) + 0..7} and {32c + 16 + 8*(l/16) + 0..7}.
    for (int s = wave; s < NTILES * KCHUNKS; s += WAVES_PER_BLOCK) {
        const int n   = s >> 2;
        const int c   = s & 3;
        const int col = n * 16 + m16;
        bf16_t* dst = &sB[s * 512 + lane * 16];
        if (col < NUM_NEG) {
            const float* row = tail_table + (long)neg_idxs[col] * EMBED;
            const int kb = 32 * c + 8 * hi16;
            #pragma unroll
            for (int j = 0; j < 8; ++j) dst[j]     = (bf16_t)row[kb + j];
            #pragma unroll
            for (int j = 0; j < 8; ++j) dst[8 + j] = (bf16_t)row[kb + 16 + j];
        } else {
            #pragma unroll
            for (int j = 0; j < 16; ++j) dst[j] = (bf16_t)0.0f;
        }
    }
    __syncthreads();

    // ---------------- Build A operand (example_vec) + fused positive dot ---
    const int tile  = blockIdx.x * WAVES_PER_BLOCK + wave;
    const int rbase = tile * 16;
    const int hrow_idx = head_idxs[rbase + m16];
    const int trow_idx = tail_idxs[rbase + m16];
    const float* hrow = head_table + (long)hrow_idx * EMBED;
    const float* trow = tail_table + (long)trow_idx * EMBED;

    v16bf A[KCHUNKS];
    float posdot = 0.0f;
    #pragma unroll
    for (int c = 0; c < KCHUNKS; ++c) {
        const int kb = 32 * c + 8 * hi16;
        #pragma unroll
        for (int j = 0; j < 8; ++j) {
            float e0 = hrow[kb + j]      + relation_vec[kb + j];
            float e1 = hrow[kb + 16 + j] + relation_vec[kb + 16 + j];
            posdot = fmaf(e0, trow[kb + j],      posdot);
            posdot = fmaf(e1, trow[kb + 16 + j], posdot);
            A[c][j]     = (bf16_t)e0;
            A[c][8 + j] = (bf16_t)e1;
        }
    }

    // ---------------- WMMA: neg_logits tile = example @ neg^T --------------
    v8f acc[NTILES];
    const v8f vzero = {};
    #pragma unroll
    for (int n = 0; n < NTILES; ++n) acc[n] = vzero;

    #pragma unroll
    for (int n = 0; n < NTILES; ++n) {
        #pragma unroll
        for (int c = 0; c < KCHUNKS; ++c) {
            v16bf b = *(const v16bf*)&sB[(n * KCHUNKS + c) * 512 + lane * 16];
            acc[n] = __builtin_amdgcn_wmma_f32_16x16x32_bf16(
                         false, A[c], false, b, (short)0, acc[n], false, false);
        }
    }

    // ---------------- Epilogue: bias + softplus, masked over valid cols ----
    // C layout: lane holds rows m = 8*(lane/16) + r (r = VGPR index), col = lane%16.
    float brow[8];
    #pragma unroll
    for (int r = 0; r < 8; ++r)
        brow[r] = bias_table[tail_idxs[rbase + 8 * hi16 + r]];

    float s = 0.0f;
    #pragma unroll
    for (int n = 0; n < NTILES; ++n) {
        const int col = n * 16 + m16;
        if (col < NUM_NEG) {
            #pragma unroll
            for (int r = 0; r < 8; ++r)
                s += softplusf(acc[n][r] + brow[r]);   // -log_sigmoid(-x)
        }
    }

    // Positive loss: complete the row dot across the lane pair (l, l^16).
    float fulldot = posdot + __shfl_xor(posdot, 16, 32);
    float bpos    = bias_table[trow_idx];
    float ploss   = softplusf(-(fulldot + bpos));      // -log_sigmoid(+x)
    if (hi16 == 0) s += ploss;                         // count each row once

    // ---------------- Deterministic reductions -----------------------------
    #pragma unroll
    for (int off = 16; off >= 1; off >>= 1)
        s += __shfl_xor(s, off, 32);
    if (lane == 0) sWave[wave] = s;
    __syncthreads();
    if (tid == 0) {
        float t = 0.0f;
        #pragma unroll
        for (int w = 0; w < WAVES_PER_BLOCK; ++w) t += sWave[w];
        block_partials[blockIdx.x] = t;
    }
}

__global__ __launch_bounds__(BLOCKS)
void ke_reduce_kernel(const float* __restrict__ partials, float* __restrict__ out)
{
    __shared__ float sm[BLOCKS];
    const int t = threadIdx.x;
    sm[t] = partials[t];
    __syncthreads();
    for (int off = BLOCKS / 2; off > 0; off >>= 1) {
        if (t < off) sm[t] += sm[t + off];
        __syncthreads();
    }
    if (t == 0) out[0] = sm[0] * (1.0f / (float)BATCH);
}

extern "C" void kernel_launch(void* const* d_in, const int* in_sizes, int n_in,
                              void* d_out, int out_size, void* d_ws, size_t ws_size,
                              hipStream_t stream)
{
    const float* head_table   = (const float*)d_in[0];
    const float* tail_table   = (const float*)d_in[1];
    const float* relation_vec = (const float*)d_in[2];
    const float* bias_table   = (const float*)d_in[3];
    const int*   head_idxs    = (const int*)d_in[4];
    const int*   tail_idxs    = (const int*)d_in[5];
    const int*   neg_idxs     = (const int*)d_in[6];
    float* partials = (float*)d_ws;   // BLOCKS floats of scratch
    float* out      = (float*)d_out;

    ke_main_kernel<<<BLOCKS, 256, 0, stream>>>(head_table, tail_table, relation_vec,
                                               bias_table, head_idxs, tail_idxs,
                                               neg_idxs, partials);
    ke_reduce_kernel<<<1, BLOCKS, 0, stream>>>(partials, out);
}
